// Conv2dLocal_1864015806835
// MI455X (gfx1250) — matrix operand order repaired and checked by reading, not compiled, not run on hardware
//
#include <hip/hip_runtime.h>
#include <hip/hip_bf16.h>

// Locally-connected conv2d: per-location GEMM A[64x576] * W[576x64] (+bias).
// Weight stream (151 MB, read once) dominates -> memory bound, ~8us floor at
// 23.3 TB/s. bf16 WMMA (16x16x32) with hi/lo fp32-emulation split (3 WMMAs per
// tile) keeps ~fp32 accuracy while the bf16 matrix rate safely exceeds the
// ~600 TFLOP/s needed to stay bandwidth-limited.
// This revision: 96-wide K staging (6 stages, 12 barriers/block instead of 36),
// packed b32 LDS fill stores, cheaper im2col index math.

typedef __attribute__((ext_vector_type(16))) __bf16 v16bf;
typedef __attribute__((ext_vector_type(8)))  float  v8f;

#define B_     64
#define CIN    64
#define HW     32
#define COUT   64
#define KTOT   576    // CIN * 3 * 3
#define KSTEP  32     // WMMA K step
#define KSTAGE 96     // K per LDS stage (3 WMMA steps)
#define NSTAGE 6      // 576 / 96
#define LOCS   1024
#define LDSROW 104    // padded row stride (bf16): 208B = 52 banks -> spread rows

union AFrag { v16bf v; uint4 q[2]; };
union BFrag { v16bf v; __bf16 e[16]; };
union P2    { __bf16 h[2]; unsigned int u; };

template<bool TWOPASS>
__global__ __launch_bounds__(128)
void conv_local_wmma(const float* __restrict__ x,
                     const float* __restrict__ wgt,
                     const float* __restrict__ bias,
                     float* __restrict__ dst) {
  __shared__ __attribute__((aligned(16))) __bf16 aHi[B_ * LDSROW];
  __shared__ __attribute__((aligned(16))) __bf16 aLo[B_ * LDSROW];

  const int loc   = blockIdx.x;          // output location h*32+w
  const int h     = loc >> 5;
  const int w     = loc & 31;
  const int tid   = threadIdx.x;
  const int lane  = tid & 31;
  const int wv    = tid >> 5;            // wave id -> N tile (16 output chans)
  const int lhalf = lane >> 4;           // 0: lanes 0-15, 1: lanes 16-31
  const int ncol  = lane & 15;
  const int oCol  = wv * 16 + ncol;      // this lane's output channel

  // Fill mapping: each thread owns two adjacent k (b32 packed stores) and
  // 8 batch rows (stride 8).
  const int kkPair = (tid & 15) * 2;
  const int bF0    = tid >> 4;

  v8f acc[4];
#pragma unroll
  for (int m = 0; m < 4; ++m)
#pragma unroll
    for (int v = 0; v < 8; ++v) acc[m][v] = 0.0f;

  // Lane's weight row: W[loc][oCol][k], contiguous in k; lane half -> K 0-15 / 16-31.
  const float* wRowBase = wgt + ((size_t)loc * COUT + oCol) * KTOT + lhalf * 16;

  for (int st = 0; st < NSTAGE; ++st) {
    const int k0 = st * KSTAGE;
    __syncthreads();  // protect LDS from previous stage's readers

    // ---- stage A (64 batches x 96 k) into LDS as bf16 hi/lo planes ----
#pragma unroll
    for (int s = 0; s < 3; ++s) {
      const int k  = k0 + s * KSTEP + kkPair;
      const int c0 = k / 9;
      const int r0 = k - 9 * c0;
      const int kh0 = r0 / 3;
      const int kw0 = r0 - 3 * kh0;
      int r1 = r0 + 1, c1 = c0;
      if (r1 == 9) { r1 = 0; ++c1; }           // derive k+1 decomposition
      const int kh1 = r1 / 3;
      const int kw1 = r1 - 3 * kh1;
      const int y0 = h + kh0 - 1, x0 = w + kw0 - 1;
      const int y1 = h + kh1 - 1, x1 = w + kw1 - 1;
      const bool ok0 = ((unsigned)y0 < 32u) && ((unsigned)x0 < 32u);
      const bool ok1 = ((unsigned)y1 < 32u) && ((unsigned)x1 < 32u);
      const float* xp0 = x + ((size_t)c0 * HW + y0) * HW + x0;
      const float* xp1 = x + ((size_t)c1 * HW + y1) * HW + x1;
      const int idxBase = s * KSTEP + kkPair;  // even -> 4B-aligned b32 stores
#pragma unroll
      for (int i = 0; i < 8; ++i) {
        const int b = bF0 + 8 * i;
        const size_t boff = (size_t)b * (CIN * HW * HW);
        const float f0 = ok0 ? xp0[boff] : 0.0f;
        const float f1 = ok1 ? xp1[boff] : 0.0f;
        const __bf16 h0 = (__bf16)f0, h1 = (__bf16)f1;
        const __bf16 l0 = (__bf16)(f0 - (float)h0);
        const __bf16 l1 = (__bf16)(f1 - (float)h1);
        P2 ph, pl;
        ph.h[0] = h0; ph.h[1] = h1;
        pl.h[0] = l0; pl.h[1] = l1;
        *(unsigned int*)&aHi[b * LDSROW + idxBase] = ph.u;
        *(unsigned int*)&aLo[b * LDSROW + idxBase] = pl.u;
      }
    }

    // ---- stream B fragments (3 K-steps; each lane pulls full 64B lines) ----
    BFrag bHi[3], bLo[3];
#pragma unroll
    for (int s = 0; s < 3; ++s) {
      const float* wp = wRowBase + k0 + s * KSTEP;
      const float4 w0 = *(const float4*)(wp + 0);
      const float4 w1 = *(const float4*)(wp + 4);
      const float4 w2 = *(const float4*)(wp + 8);
      const float4 w3 = *(const float4*)(wp + 12);
      const float wf[16] = {w0.x, w0.y, w0.z, w0.w, w1.x, w1.y, w1.z, w1.w,
                            w2.x, w2.y, w2.z, w2.w, w3.x, w3.y, w3.z, w3.w};
#pragma unroll
      for (int j = 0; j < 16; ++j) {
        const __bf16 hv = (__bf16)wf[j];
        bHi[s].e[j] = hv;
        bLo[s].e[j] = (__bf16)(wf[j] - (float)hv);
      }
      if (st + 1 < NSTAGE) __builtin_prefetch(wp + KSTAGE, 0, 1);  // next stage line
    }

    __syncthreads();

    // ---- 3 K-steps x 4 M-tiles x 3 WMMAs (hi*lo + lo*hi + hi*hi) ----
#pragma unroll
    for (int s = 0; s < 3; ++s) {
      const int off0 = s * KSTEP + (lhalf ? 16 : 0);
#pragma unroll
      for (int m = 0; m < 4; ++m) {
        const __bf16* rHi = aHi + (16 * m + ncol) * LDSROW;
        const __bf16* rLo = aLo + (16 * m + ncol) * LDSROW;
        AFrag aH, aL;
        aH.q[0] = *(const uint4*)(rHi + off0);        // K off0..off0+7
        aH.q[1] = *(const uint4*)(rHi + off0 + 8);    // K off0+8..off0+15
        aL.q[0] = *(const uint4*)(rLo + off0);
        aL.q[1] = *(const uint4*)(rLo + off0 + 8);
        acc[m] = __builtin_amdgcn_wmma_f32_16x16x32_bf16(false, aH.v, false, bLo[s].v,
                                                         (short)0, acc[m], false, false);
        acc[m] = __builtin_amdgcn_wmma_f32_16x16x32_bf16(false, aL.v, false, bHi[s].v,
                                                         (short)0, acc[m], false, false);
        acc[m] = __builtin_amdgcn_wmma_f32_16x16x32_bf16(false, aH.v, false, bHi[s].v,
                                                         (short)0, acc[m], false, false);
      }
    }
  }

  // ---- epilogue ----
  if (TWOPASS) {
    // Write [loc][b][o]: fully coalesced 16KB per block; bias fused in pass 2.
    float* base = dst + (size_t)loc * (B_ * COUT);
#pragma unroll
    for (int m = 0; m < 4; ++m)
#pragma unroll
      for (int v = 0; v < 8; ++v) {
        const int b = 16 * m + v + 8 * lhalf;      // C/D layout: M = v + 8*lhalf
        base[b * COUT + oCol] = acc[m][v];
      }
  } else {
    const float bv = bias[oCol * LOCS + loc];
#pragma unroll
    for (int m = 0; m < 4; ++m)
#pragma unroll
      for (int v = 0; v < 8; ++v) {
        const int b = 16 * m + v + 8 * lhalf;
        dst[((size_t)b * COUT + oCol) * LOCS + loc] = acc[m][v] + bv;
      }
  }
}

// Pass 2: [loc][bo] -> [bo][loc] transpose with fused bias, both sides coalesced.
__global__ __launch_bounds__(256)
void transpose_bias_kernel(const float* __restrict__ ws,
                           const float* __restrict__ bias,
                           float* __restrict__ out) {
  __shared__ float tile[32][33];
  const int locBase = blockIdx.x * 32;
  const int boBase  = blockIdx.y * 32;
  const int tx = threadIdx.x;   // 0..31
  const int ty = threadIdx.y;   // 0..7
#pragma unroll
  for (int j = 0; j < 4; ++j) {
    const int l = ty + 8 * j;
    tile[l][tx] = ws[(size_t)(locBase + l) * 4096 + boBase + tx];
  }
  __syncthreads();
#pragma unroll
  for (int j = 0; j < 4; ++j) {
    const int r  = ty + 8 * j;
    const int bo = boBase + r;
    const int o  = bo & 63;
    out[(size_t)bo * 1024 + locBase + tx] = tile[tx][r] + bias[o * 1024 + locBase + tx];
  }
}

extern "C" void kernel_launch(void* const* d_in, const int* in_sizes, int n_in,
                              void* d_out, int out_size, void* d_ws, size_t ws_size,
                              hipStream_t stream) {
  const float* x    = (const float*)d_in[0];
  const float* wgt  = (const float*)d_in[1];
  const float* bias = (const float*)d_in[2];
  float* out = (float*)d_out;

  const size_t need = (size_t)LOCS * B_ * COUT * sizeof(float);  // 16.78 MB
  if (ws_size >= need) {
    float* ws = (float*)d_ws;
    conv_local_wmma<true><<<LOCS, 128, 0, stream>>>(x, wgt, bias, ws);
    transpose_bias_kernel<<<dim3(32, 128), dim3(32, 8), 0, stream>>>(ws, bias, out);
  } else {
    conv_local_wmma<false><<<LOCS, 128, 0, stream>>>(x, wgt, bias, out);
  }
}
